// RNN_25348896981724
// MI455X (gfx1250) — matrix operand order repaired and checked by reading, not compiled, not run on hardware
//
#include <hip/hip_runtime.h>
#include <hip/hip_bf16.h>
#include <math.h>

typedef __attribute__((ext_vector_type(16))) _Float16 v16h;
typedef __attribute__((ext_vector_type(8)))  float    v8f;

#define B_   512
#define T_   2048
#define IN_  32
#define S_   16
#define N_   128
#define OUT_ 8
#define H_   64
#define DT_  0.01f

// ---------------- CDNA5 WMMA helpers ----------------

__device__ inline v8f wmma16(v16h a, v16h b, v8f c) {
  // D = A(16x32 f16) x B(32x16 f16) + C(16x16 f32)
  return __builtin_amdgcn_wmma_f32_16x16x32_f16(false, a, false, b, (short)0, c,
                                                false, false);
}

__device__ inline void lds_fence() { asm volatile("s_wait_dscnt 0x0" ::: "memory"); }

// A-matrix (16x32 f16) K-offset of packed element e for lane-half h (ISA 7.12.2)
__device__ inline int a_kl(int e, int h) {
  int j = e >> 1, p = e & 1;
  return ((j & 4) ? 16 : 0) + ((j & 3) << 1) + p + (h << 3);
}

// Build a B fragment (one 32x16 K,N window) from row-major f32 W[Kdim][Ndim].
// B layout: lane = N column (mod 16), elements follow the A-matrix K order (a_kl).
__device__ inline v16h bfrag_g(const float* W, int Kdim, int Ndim, int kt, int nt, int lane) {
  v16h r;
  int n = nt * 16 + (lane & 15);
  int koff = kt * 32;
#pragma unroll
  for (int e = 0; e < 16; ++e) {
    int k = koff + a_kl(e, lane >> 4);
    float v = (k < Kdim && n < Ndim) ? W[k * Ndim + n] : 0.0f;
    r[e] = (_Float16)v;
  }
  return r;
}

// Build an A fragment (16 x 32 window kt) from row-major f32 src[16][ld]
__device__ inline v16h afrag_f32(const float* src, int ld, int Kdim, int kt, int lane) {
  v16h r;
  int m = lane & 15, h = lane >> 4;
#pragma unroll
  for (int e = 0; e < 16; ++e) {
    int k = kt * 32 + a_kl(e, h);
    r[e] = (_Float16)((k < Kdim) ? src[m * ld + k] : 0.0f);
  }
  return r;
}

__device__ inline v16h afrag_f16(const _Float16* src, int ld, int kt, int lane) {
  v16h r;
  int m = lane & 15, h = lane >> 4;
#pragma unroll
  for (int e = 0; e < 16; ++e) {
    int k = kt * 32 + a_kl(e, h);
    r[e] = src[m * ld + k];
  }
  return r;
}

// ---------------- RNN scan kernel ----------------

enum { F_CV = 0, F_DVY = 8, F_A = 16, F_BW = 17, F_BY = 21, F_CU = 22, F_DUW = 23,
       F_DUY = 27, F_NUM = 28 };

__global__ __launch_bounds__(128) void rnn_scan_kernel(
    const float* __restrict__ obs,  const float* __restrict__ x0,
    const float* __restrict__ A_T,  const float* __restrict__ Bw_T,
    const float* __restrict__ By_T, const float* __restrict__ Cv_T,
    const float* __restrict__ Dvy_T,const float* __restrict__ Cu_T,
    const float* __restrict__ Duw_T,const float* __restrict__ Duy_T,
    const float* __restrict__ log_stds,
    float* __restrict__ out_act, float* __restrict__ out_xf)
{
  __shared__ __align__(32) _Float16 wfrag[F_NUM][32][16];  // fragmentized weights
  __shared__ _Float16 w_lds[4][16][N_];                    // per-wave w transpose staging
  __shared__ float    x_lds[4][16][S_];                    // per-wave state (f32)
  __shared__ float    xt_lds[4][16][S_];                   // per-wave RK4 midpoint

  const int lane = threadIdx.x & 31;
  const int wv   = threadIdx.x >> 5;
  const int h    = lane >> 4;
  const int n    = lane & 15;

  // One-time: fragmentize all weight matrices into LDS (shared by 4 waves).
  {
    const float* Wp[F_NUM]; int Kd[F_NUM], Nd[F_NUM], Kt[F_NUM], Nt[F_NUM];
    for (int i = 0; i < 8; ++i) { Wp[F_CV+i]=Cv_T;  Kd[F_CV+i]=S_;  Nd[F_CV+i]=N_;  Kt[F_CV+i]=0; Nt[F_CV+i]=i; }
    for (int i = 0; i < 8; ++i) { Wp[F_DVY+i]=Dvy_T;Kd[F_DVY+i]=IN_;Nd[F_DVY+i]=N_; Kt[F_DVY+i]=0;Nt[F_DVY+i]=i; }
    Wp[F_A]=A_T;   Kd[F_A]=S_;   Nd[F_A]=S_;    Kt[F_A]=0;  Nt[F_A]=0;
    for (int i = 0; i < 4; ++i) { Wp[F_BW+i]=Bw_T;  Kd[F_BW+i]=N_;  Nd[F_BW+i]=S_;  Kt[F_BW+i]=i; Nt[F_BW+i]=0; }
    Wp[F_BY]=By_T; Kd[F_BY]=IN_; Nd[F_BY]=S_;   Kt[F_BY]=0; Nt[F_BY]=0;
    Wp[F_CU]=Cu_T; Kd[F_CU]=S_;  Nd[F_CU]=OUT_; Kt[F_CU]=0; Nt[F_CU]=0;
    for (int i = 0; i < 4; ++i) { Wp[F_DUW+i]=Duw_T; Kd[F_DUW+i]=N_; Nd[F_DUW+i]=OUT_; Kt[F_DUW+i]=i; Nt[F_DUW+i]=0; }
    Wp[F_DUY]=Duy_T; Kd[F_DUY]=IN_; Nd[F_DUY]=OUT_; Kt[F_DUY]=0; Nt[F_DUY]=0;
    for (int f = wv; f < F_NUM; f += 4) {
      v16h r = bfrag_g(Wp[f], Kd[f], Nd[f], Kt[f], Nt[f], lane);
      *(v16h*)&wfrag[f][lane][0] = r;
    }
  }
  __syncthreads();

#define WFR(i) (*(const v16h*)&wfrag[(i)][lane][0])

  const int bt = blockIdx.x * 4 + wv;          // 16-row batch tile
  const float ls = (n >= OUT_) ? log_stds[n - OUT_] : 0.0f;

  // x state: f32, kept in D-fragment layout (lane = col n, rows r + 8h) + mirrored in LDS
  v8f xD;
#pragma unroll
  for (int r = 0; r < 8; ++r) {
    int m = r + 8 * h;
    float v = x0[(bt * 16 + m) * S_ + n];
    xD[r] = v;
    x_lds[wv][m][n] = v;
  }
  lds_fence();

  const v8f zero = {};

  for (int t = 0; t < T_; ++t) {
    // y A-fragment straight from global obs (f32 -> f16)
    v16h yA;
    {
      const float* yrow = obs + ((size_t)(bt * 16 + n) * T_ + t) * IN_;
#pragma unroll
      for (int e = 0; e < 16; ++e) yA[e] = (_Float16)yrow[a_kl(e, h)];
      __builtin_prefetch(yrow + IN_, 0, 1);    // speculative prefetch of next step's row
    }

    // Per-step y contributions (reused by all 4 RK4 substeps)
    v8f dvy[8];
#pragma unroll
    for (int nt = 0; nt < 8; ++nt) dvy[nt] = wmma16(yA, WFR(F_DVY + nt), zero);
    v8f ybase = wmma16(yA, WFR(F_BY), zero);   // y @ By_T
    v8f uacc  = wmma16(yA, WFR(F_DUY), zero);  // y @ Duy_T (cols 8..15 stay 0)

    v16h wA[4];
    auto compute_w = [&](v16h xa) {            // w = tanh(x@Cv_T + y@Dvy_T), as A-frags
#pragma unroll
      for (int nt = 0; nt < 8; ++nt) {
        v8f acc = wmma16(xa, WFR(F_CV + nt), dvy[nt]);
#pragma unroll
        for (int r = 0; r < 8; ++r)
          w_lds[wv][r + 8 * h][nt * 16 + n] = (_Float16)tanhf(acc[r]);
      }
      lds_fence();
#pragma unroll
      for (int kt = 0; kt < 4; ++kt) wA[kt] = afrag_f16(&w_lds[wv][0][0], N_, kt, lane);
    };
    auto xdot = [&](v16h xa) -> v8f {          // x@A_T + w@Bw_T + y@By_T
      v8f acc = wmma16(xa, WFR(F_A), ybase);
#pragma unroll
      for (int kt = 0; kt < 4; ++kt) acc = wmma16(wA[kt], WFR(F_BW + kt), acc);
      return acc;
    };

    v16h xA = afrag_f32(&x_lds[wv][0][0], S_, S_, 0, lane);
    compute_w(xA);

    // action head: u = x@Cu_T + w@Duw_T + y@Duy_T ; emit [actions | log_stds]
    uacc = wmma16(xA, WFR(F_CU), uacc);
#pragma unroll
    for (int kt = 0; kt < 4; ++kt) uacc = wmma16(wA[kt], WFR(F_DUW + kt), uacc);
#pragma unroll
    for (int r = 0; r < 8; ++r) {
      int m = r + 8 * h;
      out_act[((size_t)(bt * 16 + m) * T_ + t) * 16 + n] = (n < OUT_) ? uacc[r] : ls;
    }

    v8f k1 = xdot(xA);

    auto midA = [&](v8f k, float c) -> v16h {  // stage x + c*k, rebuild as A-frag
#pragma unroll
      for (int r = 0; r < 8; ++r) xt_lds[wv][r + 8 * h][n] = xD[r] + c * k[r];
      lds_fence();
      return afrag_f32(&xt_lds[wv][0][0], S_, S_, 0, lane);
    };

    v16h x2A = midA(k1, 0.5f * DT_);
    compute_w(x2A);
    v8f k2 = xdot(x2A);

    v16h x3A = midA(k2, 0.5f * DT_);
    compute_w(x3A);
    v8f k3 = xdot(x3A);

    v16h x4A = midA(k3, DT_);
    compute_w(x4A);
    v8f k4 = xdot(x4A);

#pragma unroll
    for (int r = 0; r < 8; ++r) {
      xD[r] += (DT_ / 6.0f) * (k1[r] + 2.0f * k2[r] + 2.0f * k3[r] + k4[r]);
      x_lds[wv][r + 8 * h][n] = xD[r];
    }
    lds_fence();
  }

#pragma unroll
  for (int r = 0; r < 8; ++r)
    out_xf[(bt * 16 + r + 8 * h) * S_ + n] = xD[r];
#undef WFR
}

// ---------------- value-baseline MLP kernel (memory-bound) ----------------

__global__ __launch_bounds__(256) void value_mlp_kernel(
    const float* __restrict__ obs,
    const float* __restrict__ W1, const float* __restrict__ b1,
    const float* __restrict__ W2, const float* __restrict__ b2,
    const float* __restrict__ W3, const float* __restrict__ b3,
    float* __restrict__ value)
{
  __shared__ _Float16 h1_lds[8][16][H_];
  __shared__ _Float16 h2_lds[8][16][H_];
  const int lane = threadIdx.x & 31;
  const int wv   = threadIdx.x >> 5;
  const int h    = lane >> 4, n = lane & 15;

  // weight fragments live in registers for the whole grid-stride loop
  v16h W1B[4], W2B[2][4], W3B[2];
#pragma unroll
  for (int nt = 0; nt < 4; ++nt) W1B[nt] = bfrag_g(W1, IN_, H_, 0, nt, lane);
#pragma unroll
  for (int kt = 0; kt < 2; ++kt)
#pragma unroll
    for (int nt = 0; nt < 4; ++nt) W2B[kt][nt] = bfrag_g(W2, H_, H_, kt, nt, lane);
#pragma unroll
  for (int kt = 0; kt < 2; ++kt) W3B[kt] = bfrag_g(W3, H_, 1, kt, 0, lane);
  float b1v[4], b2v[4];
#pragma unroll
  for (int nt = 0; nt < 4; ++nt) { b1v[nt] = b1[nt * 16 + n]; b2v[nt] = b2[nt * 16 + n]; }
  const float b3s = b3[0];

  const int ntiles = (B_ * T_) / 16;
  const int gw = blockIdx.x * 8 + wv;
  const int nw = gridDim.x * 8;
  for (int tile = gw; tile < ntiles; tile += nw) {
    const int r0 = tile * 16;
    v16h yA;
    {
      const float* yrow = obs + (size_t)(r0 + n) * IN_;
#pragma unroll
      for (int e = 0; e < 16; ++e) yA[e] = (_Float16)yrow[a_kl(e, h)];
    }
    // layer 1: tanh(obs @ W1 + b1)
#pragma unroll
    for (int nt = 0; nt < 4; ++nt) {
      v8f acc;
#pragma unroll
      for (int r = 0; r < 8; ++r) acc[r] = b1v[nt];
      acc = wmma16(yA, W1B[nt], acc);
#pragma unroll
      for (int r = 0; r < 8; ++r) h1_lds[wv][r + 8 * h][nt * 16 + n] = (_Float16)tanhf(acc[r]);
    }
    lds_fence();
    v16h h1A0 = afrag_f16(&h1_lds[wv][0][0], H_, 0, lane);
    v16h h1A1 = afrag_f16(&h1_lds[wv][0][0], H_, 1, lane);
    // layer 2: tanh(h @ W2 + b2)
#pragma unroll
    for (int nt = 0; nt < 4; ++nt) {
      v8f acc;
#pragma unroll
      for (int r = 0; r < 8; ++r) acc[r] = b2v[nt];
      acc = wmma16(h1A0, W2B[0][nt], acc);
      acc = wmma16(h1A1, W2B[1][nt], acc);
#pragma unroll
      for (int r = 0; r < 8; ++r) h2_lds[wv][r + 8 * h][nt * 16 + n] = (_Float16)tanhf(acc[r]);
    }
    lds_fence();
    v16h h2A0 = afrag_f16(&h2_lds[wv][0][0], H_, 0, lane);
    v16h h2A1 = afrag_f16(&h2_lds[wv][0][0], H_, 1, lane);
    // output: h @ W3 + b3 (N padded 1->16, only col 0 valid)
    v8f vacc;
#pragma unroll
    for (int r = 0; r < 8; ++r) vacc[r] = b3s;
    vacc = wmma16(h2A0, W3B[0], vacc);
    vacc = wmma16(h2A1, W3B[1], vacc);
    if (n == 0) {
#pragma unroll
      for (int r = 0; r < 8; ++r) value[r0 + r + 8 * h] = vacc[r];
    }
  }
}

// ---------------- launch ----------------

extern "C" void kernel_launch(void* const* d_in, const int* in_sizes, int n_in,
                              void* d_out, int out_size, void* d_ws, size_t ws_size,
                              hipStream_t stream) {
  const float* obs   = (const float*)d_in[0];
  const float* x0    = (const float*)d_in[1];
  const float* A_T   = (const float*)d_in[2];
  const float* Bw_T  = (const float*)d_in[3];
  const float* By_T  = (const float*)d_in[4];
  const float* Cv_T  = (const float*)d_in[5];
  const float* Dvy_T = (const float*)d_in[6];
  const float* Cu_T  = (const float*)d_in[7];
  const float* Duw_T = (const float*)d_in[8];
  const float* Duy_T = (const float*)d_in[9];
  const float* lstd  = (const float*)d_in[10];
  const float* W1    = (const float*)d_in[11];
  const float* b1    = (const float*)d_in[12];
  const float* W2    = (const float*)d_in[13];
  const float* b2    = (const float*)d_in[14];
  const float* W3    = (const float*)d_in[15];
  const float* b3    = (const float*)d_in[16];

  float* out     = (float*)d_out;
  float* out_act = out;                                  // [B,T,16] actions|log_stds
  float* out_xf  = out + (size_t)B_ * T_ * 16;           // [B,16]   x_final
  float* out_val = out_xf + (size_t)B_ * S_;             // [B*T]    value

  value_mlp_kernel<<<2048, 256, 0, stream>>>(obs, W1, b1, W2, b2, W3, b3, out_val);
  rnn_scan_kernel<<<B_ / 64, 128, 0, stream>>>(obs, x0, A_T, Bw_T, By_T, Cv_T, Dvy_T,
                                               Cu_T, Duw_T, Duy_T, lstd, out_act, out_xf);
}
